// GAT_28303834480677
// MI455X (gfx1250) — matrix operand order repaired and checked by reading, not compiled, not run on hardware
//
#include <hip/hip_runtime.h>

#define HID 128
#define NEG_SLOPE 0.2f

typedef __attribute__((ext_vector_type(16))) __bf16 v16bf;
typedef __attribute__((ext_vector_type(8)))  __bf16 v8bf;
typedef __attribute__((ext_vector_type(8)))  float  v8f;

// round-to-nearest-even f32 -> bf16
static __device__ inline __bf16 f2bf(float f) {
  union { float f; unsigned u; } x; x.f = f;
  unsigned r = x.u + 0x7FFFu + ((x.u >> 16) & 1u);
  unsigned short h = (unsigned short)(r >> 16);
  union { unsigned short s; __bf16 b; } y; y.s = h;
  return y.b;
}

static __device__ inline void atomicMaxF(float* a, float v) {
  if (v >= 0.0f) atomicMax((int*)a, __float_as_int(v));
  else           atomicMin((unsigned int*)a, (unsigned int)__float_as_int(v));
}

// ---------------------------------------------------------------------------
// f32 -> bf16 bulk convert (activations)
// ---------------------------------------------------------------------------
__global__ void cvt_f32_bf16(const float* __restrict__ in,
                             __bf16* __restrict__ out, long n) {
  long i = blockIdx.x * (long)blockDim.x + threadIdx.x;
  long s = gridDim.x * (long)blockDim.x;
  for (; i < n; i += s) out[i] = f2bf(in[i]);
}

// ---------------------------------------------------------------------------
// Pack W[K,N] (f32 row-major) into a contiguous-k 32x16 bf16 B-fragment layout (same k order as A):
// flat index = ((kt*tilesN + tn)*32 + lane)*16 + e
// holds W[(kt*32 + (lane>>4)*16 + e)*N + tn*16 + (lane&15)]
// so a wave's B fragment is one contiguous 32B load per lane.
// ---------------------------------------------------------------------------
__global__ void repack_w(const float* __restrict__ W, __bf16* __restrict__ out,
                         int K, int N) {
  int idx = blockIdx.x * blockDim.x + threadIdx.x;
  if (idx >= K * N) return;
  int e    = idx & 15;
  int lane = (idx >> 4) & 31;
  int tile = idx >> 9;
  int tilesN = N >> 4;
  int tn = tile % tilesN;
  int kt = tile / tilesN;
  int rlo = lane & 15, hi = lane >> 4;
  int k = kt * 32 + hi * 16 + e;
  out[idx] = f2bf(W[(size_t)k * N + tn * 16 + rlo]);
}

// ---------------------------------------------------------------------------
// C[M,N] = A[M,K](bf16 row-major) * Bp(pre-packed bf16 fragments) + bias[N]
// One wave32 per 16x16 tile; K templated so the loop fully unrolls into
// K/32 straight-line v_wmma_f32_16x16x32_bf16 ops.
// A fragment: lanes 0-15 row M=lane%16, elems 0..7 -> K=k0..k0+7 (one b128),
// elems 8..15 -> K=k0+8..k0+15 (one b128); hi lanes shifted by 16.
// C/D: VGPR r -> row r (lanes 0-15) / row r+8 (lanes 16-31), col = lane%16.
// ---------------------------------------------------------------------------
template <int K>
__global__ void gemm_bias_wmma(const __bf16* __restrict__ A,
                               const __bf16* __restrict__ Bp,
                               const float* __restrict__ bias,
                               float* __restrict__ C, int M, int N) {
  int wave = (int)((blockIdx.x * blockDim.x + threadIdx.x) >> 5);
  int lane = threadIdx.x & 31;
  int tilesN = N >> 4;
  int tiles  = (M >> 4) * tilesN;
  if (wave >= tiles) return;              // wave-uniform: EXEC all-ones inside
  int tm = wave / tilesN;
  int tn = wave - tm * tilesN;
  int rlo = lane & 15;
  int hi  = lane >> 4;

  const __bf16* Arow = A + (size_t)(tm * 16 + rlo) * K + hi * 16;
  const __bf16* Bl   = Bp + (size_t)tn * 512 + lane * 16;
  int col = tn * 16 + rlo;

  v8f acc = {};
#pragma unroll
  for (int kt = 0; kt < K / 32; ++kt) {
    const __bf16* ap = Arow + kt * 32;
    v8bf a0 = *(const v8bf*)ap;           // K = k0 .. k0+7   (hi half offset)
    v8bf a1 = *(const v8bf*)(ap + 8);     // K = k0+8 .. k0+15
    v16bf a = __builtin_shufflevector(a0, a1, 0, 1, 2, 3, 4, 5, 6, 7, 8, 9, 10,
                                      11, 12, 13, 14, 15);
    v16bf b = *(const v16bf*)(Bl + (size_t)kt * tilesN * 512);
    acc = __builtin_amdgcn_wmma_f32_16x16x32_bf16(
        false, a, false, b, (short)0, acc, false, false);
  }
#pragma unroll
  for (int r = 0; r < 8; ++r) {
    int m = tm * 16 + r + hi * 8;
    C[(size_t)m * N + col] = acc[r] + bias[col];
  }
}

// ---------------------------------------------------------------------------
__global__ void fill_f32(float* __restrict__ p, float v, long n) {
  long i = blockIdx.x * (long)blockDim.x + threadIdx.x;
  long s = gridDim.x * (long)blockDim.x;
  for (; i < n; i += s) p[i] = v;
}

__global__ void reduce_sum(const float* __restrict__ x, int n,
                           float* __restrict__ out) {
  __shared__ float sm[256];
  float acc = 0.0f;
  for (long i = blockIdx.x * (long)blockDim.x + threadIdx.x; i < n;
       i += gridDim.x * (long)blockDim.x)
    acc += x[i];
  sm[threadIdx.x] = acc;
  __syncthreads();
  for (int o = 128; o > 0; o >>= 1) {
    if ((int)threadIdx.x < o) sm[threadIdx.x] += sm[threadIdx.x + o];
    __syncthreads();
  }
  if (threadIdx.x == 0) atomicAdd(out, sm[0]);
}

// ---------------------------------------------------------------------------
// Wave-per-edge: logits[e] = dot(leaky_relu(xl[src]+xr[dst]+w*We), att),
// atomic segment-max into lmax[dst]. Rows loaded as one float4 per lane.
// Self loops: e>=E -> src=dst=e-E, w = mean edge weight.
// ---------------------------------------------------------------------------
__global__ void edge_logits(const int* __restrict__ src,
                            const int* __restrict__ dst,
                            const float* __restrict__ ew,
                            const float* __restrict__ meansum, float inv_e,
                            const float* __restrict__ xl,
                            const float* __restrict__ xr,
                            const float* __restrict__ We,
                            const float* __restrict__ att,
                            float* __restrict__ logits,
                            float* __restrict__ lmax,
                            int E, int Etot) {
  int wave = (int)((blockIdx.x * blockDim.x + threadIdx.x) >> 5);
  int lane = threadIdx.x & 31;
  if (wave >= Etot) return;
  int s, d; float w;
  if (wave < E) { s = src[wave]; d = dst[wave]; w = ew[wave]; }
  else          { s = d = wave - E; w = meansum[0] * inv_e; }
  float4 va = ((const float4*)(xl + (size_t)s * HID))[lane];
  float4 vb = ((const float4*)(xr + (size_t)d * HID))[lane];
  float4 vw = ((const float4*)We)[lane];
  float4 vt = ((const float4*)att)[lane];
  float m0 = va.x + vb.x + w * vw.x;
  float m1 = va.y + vb.y + w * vw.y;
  float m2 = va.z + vb.z + w * vw.z;
  float m3 = va.w + vb.w + w * vw.w;
  m0 = m0 > 0.0f ? m0 : NEG_SLOPE * m0;
  m1 = m1 > 0.0f ? m1 : NEG_SLOPE * m1;
  m2 = m2 > 0.0f ? m2 : NEG_SLOPE * m2;
  m3 = m3 > 0.0f ? m3 : NEG_SLOPE * m3;
  float acc = m0 * vt.x + m1 * vt.y + m2 * vt.z + m3 * vt.w;
#pragma unroll
  for (int o = 16; o > 0; o >>= 1) acc += __shfl_xor(acc, o, 32);
  if (lane == 0) {
    logits[wave] = acc;
    atomicMaxF(&lmax[d], acc);
  }
}

// w[e] = exp(logits[e]-lmax[dst]); denom[dst] += w  (in-place over logits)
__global__ void edge_softmax(const int* __restrict__ dst,
                             float* __restrict__ logits_w,
                             const float* __restrict__ lmax,
                             float* __restrict__ denom, int E, int Etot) {
  int e = blockIdx.x * blockDim.x + threadIdx.x;
  if (e >= Etot) return;
  int d = (e < E) ? dst[e] : (e - E);
  float w = __expf(logits_w[e] - lmax[d]);
  logits_w[e] = w;
  atomicAdd(&denom[d], w);
}

// agg[dst] += xl[src] * (w[e]/denom[dst]); wave-per-edge, float4 row loads
__global__ void edge_aggregate(const int* __restrict__ src,
                               const int* __restrict__ dst,
                               const float* __restrict__ wbuf,
                               const float* __restrict__ denom,
                               const float* __restrict__ xl,
                               float* __restrict__ agg, int E, int Etot) {
  int wave = (int)((blockIdx.x * blockDim.x + threadIdx.x) >> 5);
  int lane = threadIdx.x & 31;
  if (wave >= Etot) return;
  int s, d;
  if (wave < E) { s = src[wave]; d = dst[wave]; }
  else          { s = d = wave - E; }
  float alpha = wbuf[wave] / denom[d];
  float4 v = ((const float4*)(xl + (size_t)s * HID))[lane];
  float* out = agg + (size_t)d * HID + lane * 4;
  atomicAdd(out + 0, v.x * alpha);
  atomicAdd(out + 1, v.y * alpha);
  atomicAdd(out + 2, v.z * alpha);
  atomicAdd(out + 3, v.w * alpha);
}

__global__ void bias_relu(float* __restrict__ h, const float* __restrict__ bias,
                          long n) {
  long i = blockIdx.x * (long)blockDim.x + threadIdx.x;
  long s = gridDim.x * (long)blockDim.x;
  for (; i < n; i += s) {
    float v = h[i] + bias[i & (HID - 1)];
    h[i] = v > 0.0f ? v : 0.0f;
  }
}

// global mean-pool accumulation
__global__ void pool_add(const float* __restrict__ h,
                         const int* __restrict__ batch,
                         float* __restrict__ pool, float* __restrict__ cnt,
                         int Nn) {
  long i = blockIdx.x * (long)blockDim.x + threadIdx.x;
  long n = (long)Nn * HID;
  if (i >= n) return;
  int node = (int)(i >> 7);
  int f = (int)(i & (HID - 1));
  int g = batch[node];
  atomicAdd(&pool[(size_t)g * HID + f], h[i]);
  if (f == 0) atomicAdd(&cnt[g], 1.0f);
}

// out[g] = (pool[g]/max(cnt,1)) . Wlin + blin  (wave-per-graph)
__global__ void final_linear(const float* __restrict__ pool,
                             const float* __restrict__ cnt,
                             const float* __restrict__ Wlin,
                             const float* __restrict__ blin,
                             float* __restrict__ out, int G) {
  int wave = (int)((blockIdx.x * blockDim.x + threadIdx.x) >> 5);
  int lane = threadIdx.x & 31;
  if (wave >= G) return;
  float c = cnt[wave] > 1.0f ? cnt[wave] : 1.0f;
  float4 p = ((const float4*)(pool + (size_t)wave * HID))[lane];
  float4 wl = ((const float4*)Wlin)[lane];
  float acc = (p.x * wl.x + p.y * wl.y + p.z * wl.z + p.w * wl.w) / c;
#pragma unroll
  for (int o = 16; o > 0; o >>= 1) acc += __shfl_xor(acc, o, 32);
  if (lane == 0) out[wave] = acc + blin[0];
}

// ---------------------------------------------------------------------------
extern "C" void kernel_launch(void* const* d_in, const int* in_sizes, int n_in,
                              void* d_out, int out_size, void* d_ws,
                              size_t ws_size, hipStream_t stream) {
  (void)n_in; (void)ws_size;
  const float* x    = (const float*)d_in[0];
  const int*   ei   = (const int*)d_in[1];
  const float* ew   = (const float*)d_in[2];
  const int*   bat  = (const int*)d_in[3];
  const float* W1l  = (const float*)d_in[4];
  const float* b1l  = (const float*)d_in[5];
  const float* W1r  = (const float*)d_in[6];
  const float* b1r  = (const float*)d_in[7];
  const float* We1  = (const float*)d_in[8];
  const float* att1 = (const float*)d_in[9];
  const float* bi1  = (const float*)d_in[10];
  const float* W2l  = (const float*)d_in[11];
  const float* b2l  = (const float*)d_in[12];
  const float* W2r  = (const float*)d_in[13];
  const float* b2r  = (const float*)d_in[14];
  const float* We2  = (const float*)d_in[15];
  const float* att2 = (const float*)d_in[16];
  const float* bi2  = (const float*)d_in[17];
  const float* Wlin = (const float*)d_in[18];
  const float* blin = (const float*)d_in[19];
  float* out = (float*)d_out;

  const int Nn   = in_sizes[3];          // 50000 nodes
  const int E    = in_sizes[2];          // 800000 edges
  const int Fin  = in_sizes[0] / Nn;     // 64
  const int H    = in_sizes[5];          // 128 (== HID)
  const int Fout = in_sizes[19];         // 1
  const int G    = out_size / Fout;      // 64 graphs
  const int Etot = E + Nn;               // + self loops
  const int* srcp = ei;
  const int* dstp = ei + E;

  // ---- workspace layout: bf16 region first (32B-aligned loads), then f32
  char* base = (char*)d_ws;
  __bf16* XA   = (__bf16*)base;                 base += (size_t)Nn * H * 2;
  __bf16* W1LP = (__bf16*)base;                 base += (size_t)Fin * H * 2;
  __bf16* W1RP = (__bf16*)base;                 base += (size_t)Fin * H * 2;
  __bf16* W2LP = (__bf16*)base;                 base += (size_t)H * H * 2;
  __bf16* W2RP = (__bf16*)base;                 base += (size_t)H * H * 2;
  float* XL   = (float*)base;                   base += (size_t)Nn * H * 4;
  float* XR   = (float*)base;                   base += (size_t)Nn * H * 4;
  float* H1   = (float*)base;                   base += (size_t)Nn * H * 4;
  float* H2   = (float*)base;                   base += (size_t)Nn * H * 4;
  float* LOGW = (float*)base;                   base += (size_t)Etot * 4;
  float* LMAX = (float*)base;                   base += (size_t)Nn * 4;
  float* DEN  = (float*)base;                   base += (size_t)Nn * 4;
  float* POOL = (float*)base;                   base += (size_t)G * H * 4;
  float* CNT  = (float*)base;                   base += (size_t)G * 4;
  float* MSUM = (float*)base;                   base += 4;

  // mean edge weight
  fill_f32<<<1, 64, 0, stream>>>(MSUM, 0.0f, 1);
  reduce_sum<<<512, 256, 0, stream>>>(ew, E, MSUM);

  // pre-pack weights into B-fragment layout
  repack_w<<<(Fin * H + 255) / 256, 256, 0, stream>>>(W1l, W1LP, Fin, H);
  repack_w<<<(Fin * H + 255) / 256, 256, 0, stream>>>(W1r, W1RP, Fin, H);
  repack_w<<<(H * H + 255) / 256, 256, 0, stream>>>(W2l, W2LP, H, H);
  repack_w<<<(H * H + 255) / 256, 256, 0, stream>>>(W2r, W2RP, H, H);

  const int tiles   = (Nn / 16) * (H / 16);
  const int gblocks = (tiles + 7) / 8;        // 8 waves per 256-thread block
  const int eblocks = (Etot + 7) / 8;         // wave-per-edge kernels
  const int tblocks = (Etot + 255) / 256;     // thread-per-edge kernels
  const long nh = (long)Nn * H;
  const float inv_e = 1.0f / (float)E;
  const float ninf = -__builtin_inff();

  auto run_layer = [&](const __bf16* Abf, int K, const __bf16* Wlp,
                       const float* bl, const __bf16* Wrp, const float* br,
                       const float* We, const float* att, const float* bias,
                       float* outbuf) {
    if (K == 64) {
      gemm_bias_wmma<64><<<gblocks, 256, 0, stream>>>(Abf, Wlp, bl, XL, Nn, H);
      gemm_bias_wmma<64><<<gblocks, 256, 0, stream>>>(Abf, Wrp, br, XR, Nn, H);
    } else {
      gemm_bias_wmma<128><<<gblocks, 256, 0, stream>>>(Abf, Wlp, bl, XL, Nn, H);
      gemm_bias_wmma<128><<<gblocks, 256, 0, stream>>>(Abf, Wrp, br, XR, Nn, H);
    }
    fill_f32<<<256, 256, 0, stream>>>(LMAX, ninf, Nn);
    fill_f32<<<256, 256, 0, stream>>>(DEN, 0.0f, Nn);
    fill_f32<<<4096, 256, 0, stream>>>(outbuf, 0.0f, nh);
    edge_logits<<<eblocks, 256, 0, stream>>>(srcp, dstp, ew, MSUM, inv_e, XL,
                                             XR, We, att, LOGW, LMAX, E, Etot);
    edge_softmax<<<tblocks, 256, 0, stream>>>(dstp, LOGW, LMAX, DEN, E, Etot);
    edge_aggregate<<<eblocks, 256, 0, stream>>>(srcp, dstp, LOGW, DEN, XL,
                                                outbuf, E, Etot);
    bias_relu<<<4096, 256, 0, stream>>>(outbuf, bias, nh);
  };

  // layer 1: convert x -> bf16 once, run
  cvt_f32_bf16<<<2048, 256, 0, stream>>>(x, XA, (long)Nn * Fin);
  run_layer(XA, Fin, W1LP, b1l, W1RP, b1r, We1, att1, bi1, H1);

  // layer 2: convert h1 -> bf16 once, run
  cvt_f32_bf16<<<2048, 256, 0, stream>>>(H1, XA, (long)Nn * H);
  run_layer(XA, H, W2LP, b2l, W2RP, b2r, We2, att2, bi2, H2);

  // global mean pool + final linear
  fill_f32<<<32, 256, 0, stream>>>(POOL, 0.0f, (long)G * H);
  fill_f32<<<1, 64, 0, stream>>>(CNT, 0.0f, G);
  pool_add<<<(int)((nh + 255) / 256), 256, 0, stream>>>(H2, bat, POOL, CNT, Nn);
  final_linear<<<(G * 32 + 255) / 256, 256, 0, stream>>>(POOL, CNT, Wlin, blin,
                                                         out, G);
}
